// HyperRuleGNN_59330678227223
// MI455X (gfx1250) — matrix operand order repaired and checked by reading, not compiled, not run on hardware
//
#include <hip/hip_runtime.h>
#include <hip/hip_bf16.h>

typedef unsigned short u16;
typedef unsigned int   u32;
typedef __attribute__((ext_vector_type(16))) __bf16 v16bf;
typedef __attribute__((ext_vector_type(8)))  float  v8f;
typedef __attribute__((ext_vector_type(4)))  u32    u32x4;

#define FDIM 128
#define KP   136            // padded LDS row stride (halves): breaks 64-bank stride
#define WLDS (128*KP)
#define ALDS (64*KP)
#define BPT  256            // blocks per edge type

#define GF_CLIP 1
#define GF_SIG  2
#define GF_ACC  4

__device__ __forceinline__ u16 f2b(float f) {           // f32 -> bf16 RNE
  u32 u = __float_as_uint(f);
  u32 r = u + 0x7fffu + ((u >> 16) & 1u);
  return (u16)(r >> 16);
}
__device__ __forceinline__ float clip01(float v) { return fminf(fmaxf(v, 0.f), 1.f); }

union Frag { v16bf v; u32x4 q[2]; };

// ---- CDNA5 async staging: memory -> LDS, tracked by ASYNCcnt -----------------
__device__ __forceinline__ void async_b128(u32 lds_addr, u32 voff, const u16* base) {
  asm volatile("global_load_async_to_lds_b128 %0, %1, %2"
               :: "v"(lds_addr), "v"(voff), "s"(base) : "memory");
}
__device__ __forceinline__ void wait_async0() {
  asm volatile("s_wait_asynccnt 0x0" ::: "memory");
}

// issue async gather of a full 128x128 bf16 W into sW (once per block)
__device__ __forceinline__ void issue_W(const u16* wB, u16* sW, int tid) {
  for (int idx = tid; idx < 128 * 16; idx += 128) {
    int r = idx >> 4, c = idx & 15;
    async_b128((u32)(size_t)(sW + r * KP) + (u32)(c << 4),
               (u32)((r << 8) + (c << 4)), wB);
  }
}
// issue async gather of 64 A rows (dense or index-gathered) into sAbuf
__device__ __forceinline__ void issue_A(const u16* inB, const int* rowIdx,
                                        int g, int M, u16* sAbuf, int tid) {
  for (int idx = tid; idx < 64 * 16; idx += 128) {
    int r = idx >> 4, c = idx & 15;
    int grow = g * 64 + r;
    int srow = 0;                        // pad rows load row 0; never stored
    if (grow < M) srow = rowIdx ? rowIdx[grow] : grow;
    async_b128((u32)(size_t)(sAbuf + r * KP) + (u32)(c << 4),
               (u32)(((size_t)srow << 8) + (c << 4)), inB);
  }
}
// issue async gather of 64 edge-source rows (src ids staged in LDS) into sAbuf
__device__ __forceinline__ void issue_A_edges(const u16* xb, const int* sSrc,
                                              u16* sAbuf, int tid) {
  for (int idx = tid; idx < 64 * 16; idx += 128) {
    int r = idx >> 4, c = idx & 15;
    int srow = sSrc[r];
    async_b128((u32)(size_t)(sAbuf + r * KP) + (u32)(c << 4),
               (u32)(((size_t)srow << 8) + (c << 4)), xb);
  }
}

// ---- WMMA core: one wave computes a 16x128 tile, K=128 in 4 chunks of 32 ----
__device__ __forceinline__ void wmma_tile(const u16* sA_tile, const u16* sW,
                                          int lane, v8f acc[8]) {
  const int lhi = lane >> 4;   // 0/1: which K-half this lane holds
  const int lm  = lane & 15;
#pragma unroll
  for (int kc = 0; kc < 4; ++kc) {
    Frag a;
    const u32x4* ap = (const u32x4*)(sA_tile + lm * KP + kc * 32 + (lhi << 4));
    a.q[0] = ap[0];            // k = kc*32 + lhi*16 + 0..7
    a.q[1] = ap[1];            // k = kc*32 + lhi*16 + 8..15
#pragma unroll
    for (int nt = 0; nt < 8; ++nt) {
      Frag b;
      const u32x4* bp = (const u32x4*)(sW + (nt * 16 + lm) * KP + kc * 32 + (lhi << 4));
      b.q[0] = bp[0];          // k = kc*32 + lhi*16 + 0..7
      b.q[1] = bp[1];          // k = kc*32 + lhi*16 + 8..15
      acc[nt] = __builtin_amdgcn_wmma_f32_16x16x32_bf16(
          false, a.v, false, b.v, (short)0, acc[nt], false, false);
    }
  }
}

// ---- generic dense GEMM: out[M,128] = act(in[rowIdx?][128] @ W.T + bias) ----
__global__ __launch_bounds__(128) void k_gemm(
    const u16* __restrict__ inB, const int* __restrict__ rowIdx, int M,
    const u16* __restrict__ wB,
    const float* __restrict__ bias0, const float* __restrict__ bias1, int split,
    float* __restrict__ outF, u16* __restrict__ outB, int flags)
{
  __shared__ u16 sW[WLDS];
  __shared__ u16 sA[2 * ALDS];
  const int tid = threadIdx.x, lane = tid & 31, wave = tid >> 5;

  issue_W(wB, sW, tid);                                      // async, waited below
  const int groups = (M + 63) >> 6;
  const int g0 = blockIdx.x;
  if (g0 < groups) issue_A(inB, rowIdx, g0, M, sA, tid);

  int cur = 0;
  for (int g = g0; g < groups; g += gridDim.x) {
    wait_async0();                                           // per-wave ASYNCcnt
    __syncthreads();                                         // tile + W visible
    const int gn = g + gridDim.x;
    if (gn < groups) issue_A(inB, rowIdx, gn, M, sA + (1 - cur) * ALDS, tid);

    v8f acc[8];
#pragma unroll
    for (int nt = 0; nt < 8; ++nt)
#pragma unroll
      for (int v = 0; v < 8; ++v) acc[nt][v] = 0.f;
    wmma_tile(sA + cur * ALDS + wave * 16 * KP, sW, lane, acc);

    const int lhi = lane >> 4, lm = lane & 15;
    const int gr0 = g * 64 + wave * 16;
#pragma unroll
    for (int nt = 0; nt < 8; ++nt) {
      int col = nt * 16 + lm;
#pragma unroll
      for (int v = 0; v < 8; ++v) {
        int grow = gr0 + v + (lhi << 3);
        if (grow < M) {
          size_t o = (size_t)grow * FDIM + col;
          float val = acc[nt][v];
          if (flags & GF_ACC) val += outF[o];
          if (bias0) {
            const float* bp = (bias1 && grow >= split) ? bias1 : bias0;
            val += bp[col];
          }
          if (flags & GF_SIG)  val = 1.f / (1.f + __expf(-val));
          if (flags & GF_CLIP) val = clip01(val);
          outF[o] = val;
          if (outB) outB[o] = f2b(val);
        }
      }
    }
    __syncthreads();                                         // done reading cur buf
    cur ^= 1;
  }
}

// ---- edge message pass: acc[dst] += x[src] @ B[type].T  (edges bucketed by type) ----
__global__ __launch_bounds__(128) void k_edges(
    const u16* __restrict__ xb, const int* __restrict__ perm,
    const int* __restrict__ esrc, const int* __restrict__ edst,
    const int* __restrict__ toff, const u16* __restrict__ Ball,
    float* __restrict__ acc)
{
  __shared__ u16 sW[WLDS];
  __shared__ u16 sA[2 * ALDS];
  __shared__ int sSrc[2][64];
  __shared__ int sDst[2][64];
  const int tid = threadIdx.x, lane = tid & 31, wave = tid >> 5;
  const int t = blockIdx.x / BPT, bi = blockIdx.x % BPT;
  const int e0 = toff[t], e1 = toff[t + 1];
  const u16* wB = Ball + (size_t)t * (FDIM * FDIM);

  issue_W(wB, sW, tid);                                      // async, waited below
  const int groups = ((e1 - e0) + 63) >> 6;

  if (bi < groups) {                                         // prologue: meta + A(g0)
    if (tid < 64) {
      int epos = e0 + bi * 64 + tid;
      int s = 0, d = -1;
      if (epos < e1) { int e = perm[epos]; s = esrc[e]; d = edst[e]; }
      sSrc[0][tid] = s; sDst[0][tid] = d;
    }
    __syncthreads();
    issue_A_edges(xb, sSrc[0], sA, tid);
  }

  int cur = 0;
  for (int g = bi; g < groups; g += BPT) {
    const int gn = g + BPT;
    if (gn < groups && tid < 64) {                           // meta for next group
      int epos = e0 + gn * 64 + tid;
      int s = 0, d = -1;
      if (epos < e1) { int e = perm[epos]; s = esrc[e]; d = edst[e]; }
      sSrc[1 - cur][tid] = s; sDst[1 - cur][tid] = d;
    }
    wait_async0();                                           // A(g) + W landed
    __syncthreads();
    if (gn < groups) issue_A_edges(xb, sSrc[1 - cur], sA + (1 - cur) * ALDS, tid);

    v8f a8[8];
#pragma unroll
    for (int nt = 0; nt < 8; ++nt)
#pragma unroll
      for (int v = 0; v < 8; ++v) a8[nt][v] = 0.f;
    wmma_tile(sA + cur * ALDS + wave * 16 * KP, sW, lane, a8);

    const int lhi = lane >> 4, lm = lane & 15;
#pragma unroll
    for (int nt = 0; nt < 8; ++nt) {
      int col = nt * 16 + lm;
#pragma unroll
      for (int v = 0; v < 8; ++v) {
        int d = sDst[cur][wave * 16 + v + (lhi << 3)];
        if (d >= 0) unsafeAtomicAdd(&acc[(size_t)d * FDIM + col], a8[nt][v]);
      }
    }
    __syncthreads();                                         // done with cur buffers
    cur ^= 1;
  }
}

// ---- small helper kernels ----
__global__ void k_cvt(const float* __restrict__ s, u16* __restrict__ d, long n) {
  long i = (long)blockIdx.x * blockDim.x + threadIdx.x;
  if (i < n) d[i] = f2b(s[i]);
}
__global__ void k_split_gate(const float* __restrict__ wg, u16* __restrict__ L,
                             u16* __restrict__ R) {   // [128,256] -> two [128,128]
  int i = blockIdx.x * blockDim.x + threadIdx.x;
  if (i < 128 * 256) {
    int row = i >> 8, col = i & 255;
    u16 v = f2b(wg[i]);
    if (col < 128) L[row * 128 + col] = v; else R[row * 128 + col - 128] = v;
  }
}
__global__ void k_zero_ints(int* p, int n) {
  int i = blockIdx.x * blockDim.x + threadIdx.x;
  if (i < n) p[i] = 0;
}
__global__ void k_hist(const int* __restrict__ et, int* __restrict__ counts, int E) {
  int i = blockIdx.x * blockDim.x + threadIdx.x;
  if (i < E) atomicAdd(&counts[et[i]], 1);
}
__global__ void k_scan(const int* __restrict__ counts, int* __restrict__ toff,
                       int* __restrict__ cursor, int T) {
  if (threadIdx.x == 0 && blockIdx.x == 0) {
    toff[0] = 0;
    for (int t = 0; t < T; ++t) { toff[t + 1] = toff[t] + counts[t]; cursor[t] = toff[t]; }
  }
}
__global__ void k_perm(const int* __restrict__ et, int* __restrict__ cursor,
                       int* __restrict__ perm, int E) {
  int i = blockIdx.x * blockDim.x + threadIdx.x;
  if (i < E) { int pos = atomicAdd(&cursor[et[i]], 1); perm[pos] = i; }
}
__global__ void k_clip_bf16(float* __restrict__ a, u16* __restrict__ b, long n) {
  long i = (long)blockIdx.x * blockDim.x + threadIdx.x;
  if (i < n) { float v = clip01(a[i]); a[i] = v; b[i] = f2b(v); }
}
__global__ __launch_bounds__(128) void k_segmean(
    const float* __restrict__ hf, const int* __restrict__ ptr,
    const int* __restrict__ sid, float* __restrict__ sF, u16* __restrict__ sB, int H) {
  int wave = threadIdx.x >> 5, L = threadIdx.x & 31;
  for (int h = blockIdx.x * 4 + wave; h < H; h += gridDim.x * 4) {
    int p0 = ptr[h], p1 = ptr[h + 1];
    float s0 = 0.f, s1 = 0.f, s2 = 0.f, s3 = 0.f;
    for (int p = p0; p < p1; ++p) {
      const float* r = hf + (size_t)sid[p] * FDIM;
      s0 += r[L]; s1 += r[L + 32]; s2 += r[L + 64]; s3 += r[L + 96];
    }
    float inv = (p1 > p0) ? 1.f / (float)(p1 - p0) : 0.f;
    size_t o = (size_t)h * FDIM;
    sF[o + L]      = s0 * inv; sB[o + L]      = f2b(s0 * inv);
    sF[o + L + 32] = s1 * inv; sB[o + L + 32] = f2b(s1 * inv);
    sF[o + L + 64] = s2 * inv; sB[o + L + 64] = f2b(s2 * inv);
    sF[o + L + 96] = s3 * inv; sB[o + L + 96] = f2b(s3 * inv);
  }
}
__global__ void k_gate_apply(const float* __restrict__ h2, const int* __restrict__ tgt,
                             const float* __restrict__ w, const float* __restrict__ gate,
                             const float* __restrict__ msg, u16* __restrict__ xb, int H) {
  int i = blockIdx.x * blockDim.x + threadIdx.x;
  if (i < H * FDIM) {
    int h = i >> 7, c = i & 127;
    int tr = tgt[h];
    float v = h2[(size_t)tr * FDIM + c] + w[h] * gate[i] * msg[i];
    xb[(size_t)tr * FDIM + c] = f2b(v);
  }
}

extern "C" void kernel_launch(void* const* d_in, const int* in_sizes, int n_in,
                              void* d_out, int out_size, void* d_ws, size_t ws_size,
                              hipStream_t stream) {
  const float* x    = (const float*)d_in[0];
  const int*   eidx = (const int*)d_in[1];
  const int*   et   = (const int*)d_in[2];
  const int*   hptr = (const int*)d_in[3];
  const int*   hsrc = (const int*)d_in[4];
  const int*   htgt = (const int*)d_in[5];
  const float* hw   = (const float*)d_in[6];
  const float* A1   = (const float*)d_in[7];
  const float* B1   = (const float*)d_in[8];
  const float* bs1  = (const float*)d_in[9];
  const float* bp1  = (const float*)d_in[10];
  const float* A2   = (const float*)d_in[11];
  const float* B2   = (const float*)d_in[12];
  const float* bs2  = (const float*)d_in[13];
  const float* bp2  = (const float*)d_in[14];
  const float* Wmsg = (const float*)d_in[15];
  const float* bmsg = (const float*)d_in[16];
  const float* Wupd = (const float*)d_in[17];
  const float* bupd = (const float*)d_in[18];
  const float* Wg   = (const float*)d_in[19];
  const float* bg   = (const float*)d_in[20];
  (void)n_in; (void)ws_size; (void)out_size;

  const int N = in_sizes[0] / FDIM;
  const int E = in_sizes[2];
  const int H = in_sizes[6];
  const int T = in_sizes[8] / (FDIM * FDIM);
  const int S = 40000;                       // singleton split (reference constant)
  const int* esrc = eidx;
  const int* edst = eidx + E;

  char* ws = (char*)d_ws;
  size_t off = 0;
  auto take = [&](size_t bytes) { size_t o = off; off += (bytes + 255) & ~(size_t)255; return o; };
  u16*   xb    = (u16*)  (ws + take((size_t)N * FDIM * 2));
  float* accB  = (float*)(ws + take((size_t)N * FDIM * 4));
  float* sembF = (float*)(ws + take((size_t)H * FDIM * 4));
  u16*   sembB = (u16*)  (ws + take((size_t)H * FDIM * 2));
  float* msgF  = (float*)(ws + take((size_t)H * FDIM * 4));
  u16*   msgB  = (u16*)  (ws + take((size_t)H * FDIM * 2));
  float* gateF = (float*)(ws + take((size_t)H * FDIM * 4));
  int*   perm  = (int*)  (ws + take((size_t)E * 4));
  int*   ibuf  = (int*)  (ws + take(64 * 4));
  int* counts = ibuf, *toff = ibuf + 8, *cursor = ibuf + 20;
  u16* A1b  = (u16*)(ws + take((size_t)FDIM * FDIM * 2));
  u16* B1b  = (u16*)(ws + take((size_t)T * FDIM * FDIM * 2));
  u16* A2b  = (u16*)(ws + take((size_t)FDIM * FDIM * 2));
  u16* B2b  = (u16*)(ws + take((size_t)T * FDIM * FDIM * 2));
  u16* Wmb  = (u16*)(ws + take((size_t)FDIM * FDIM * 2));
  u16* Wub  = (u16*)(ws + take((size_t)FDIM * FDIM * 2));
  u16* WgL  = (u16*)(ws + take((size_t)FDIM * FDIM * 2));
  u16* WgR  = (u16*)(ws + take((size_t)FDIM * FDIM * 2));

  auto cdiv = [](long a, long b) { return (int)((a + b - 1) / b); };
  const long NF = (long)N * FDIM;
  const int  WW = FDIM * FDIM;

  // conversions
  k_cvt<<<cdiv(NF, 256), 256, 0, stream>>>(x, xb, NF);
  k_cvt<<<cdiv(WW, 256), 256, 0, stream>>>(A1, A1b, WW);
  k_cvt<<<cdiv((long)T * WW, 256), 256, 0, stream>>>(B1, B1b, (long)T * WW);
  k_cvt<<<cdiv(WW, 256), 256, 0, stream>>>(A2, A2b, WW);
  k_cvt<<<cdiv((long)T * WW, 256), 256, 0, stream>>>(B2, B2b, (long)T * WW);
  k_cvt<<<cdiv(WW, 256), 256, 0, stream>>>(Wmsg, Wmb, WW);
  k_cvt<<<cdiv(WW, 256), 256, 0, stream>>>(Wupd, Wub, WW);
  k_split_gate<<<cdiv(128 * 256, 256), 256, 0, stream>>>(Wg, WgL, WgR);

  // bucket edges by type
  k_zero_ints<<<1, 64, 0, stream>>>(ibuf, 64);
  k_hist<<<cdiv(E, 256), 256, 0, stream>>>(et, counts, E);
  k_scan<<<1, 1, 0, stream>>>(counts, toff, cursor, T);
  k_perm<<<cdiv(E, 256), 256, 0, stream>>>(et, cursor, perm, E);

  const int gN = cdiv(N, 64), gH = cdiv(H, 64);

  // conv layer 1
  k_gemm<<<gN, 128, 0, stream>>>(xb, nullptr, N, A1b, bs1, bp1, S, accB, nullptr, 0);
  k_edges<<<T * BPT, 128, 0, stream>>>(xb, perm, esrc, edst, toff, B1b, accB);
  k_clip_bf16<<<cdiv(NF, 256), 256, 0, stream>>>(accB, xb, NF);

  // conv layer 2
  k_gemm<<<gN, 128, 0, stream>>>(xb, nullptr, N, A2b, bs2, bp2, S, accB, nullptr, 0);
  k_edges<<<T * BPT, 128, 0, stream>>>(xb, perm, esrc, edst, toff, B2b, accB);
  k_clip_bf16<<<cdiv(NF, 256), 256, 0, stream>>>(accB, xb, NF);

  // rule layer
  k_segmean<<<cdiv(H, 4), 128, 0, stream>>>(accB, hptr, hsrc, sembF, sembB, H);
  k_gemm<<<gH, 128, 0, stream>>>(sembB, nullptr, H, Wmb, bmsg, nullptr, H, msgF, msgB, 0);
  k_gemm<<<gH, 128, 0, stream>>>(xb, htgt, H, WgL, nullptr, nullptr, H, gateF, nullptr, 0);
  k_gemm<<<gH, 128, 0, stream>>>(msgB, nullptr, H, WgR, bg, nullptr, H, gateF, nullptr,
                                 GF_ACC | GF_SIG);
  k_gate_apply<<<cdiv((long)H * FDIM, 256), 256, 0, stream>>>(accB, htgt, hw, gateF, msgF, xb, H);
  k_gemm<<<gN, 128, 0, stream>>>(xb, nullptr, N, Wub, bupd, nullptr, N,
                                 (float*)d_out, nullptr, GF_CLIP);
}
